// AdamLayer_37022618091926
// MI455X (gfx1250) — matrix operand order repaired and checked by reading, not compiled, not run on hardware
//
#include <hip/hip_runtime.h>
#include <hip/hip_bf16.h>

// ---------------------------------------------------------------------------
// MoE(top-2 of 8) FFN + Adam + LayerNorm, routed (compute only selected
// experts), bf16 WMMA (v_wmma_f32_16x16x32_bf16) for both GEMMs.
// Shapes: B=2 S=2048 -> 4096 tokens, D=512, H=2048, E=8, K=2.
// M-tile = 32 tokens (2 WMMA row-tiles) so every B fragment is reused twice,
// halving L2 weight traffic vs a 16-token tile.
// ---------------------------------------------------------------------------

typedef __attribute__((ext_vector_type(16))) __bf16 v16bf;
typedef __attribute__((ext_vector_type(8)))  float  v8f;

#define N_TOK 4096
#define DDIM  512
#define HDIM  2048
#define NEXP  8
#define NELEM (N_TOK * DDIM)   // 2,097,152

#define MT     32              // tokens per FFN block (2 WMMA m-tiles)
#define TILES  (N_TOK / MT)    // 128 tile slots per expert
#define HCHUNK 256             // H chunk held in LDS per pass
#define XS_STR (DDIM + 8)      // padded LDS strides (bank-conflict free)
#define HS_STR (HCHUNK + 8)

#define MU_C    0.7f
#define G1_C    1.0f
#define G2_C    1.0f
#define BETA1_C 0.9f
#define BETA2_C 0.999f
#define EPSA_C  1e-8f
#define EPSLN_C 1e-5f

// ---------------------------------------------------------------------------
// Build a 16x32 bf16 A-fragment from a row-major LDS tile.
// Layout used here (not the ISA table order): lanes 0-15 & 16-31 both hold M=lane&15;
// half = lane>>4 selects K subset: VGPR j -> k = 16*half + 2j (+0/1)
// (contiguous order, matching the packed B fragments).
// ---------------------------------------------------------------------------
__device__ __forceinline__ v16bf load_a16(const __bf16* base, int stride,
                                          int m, int k0, int half) {
  v16bf a;
  const __bf16* row = base + m * stride + k0;
#pragma unroll
  for (int j = 0; j < 8; ++j) {
    int kp = 16 * half + 2 * j;
    a[2 * j]     = row[kp];
    a[2 * j + 1] = row[kp + 1];
  }
  return a;
}

__device__ __forceinline__ v8f wmma_bf16(v16bf a, v16bf b, v8f c) {
  return __builtin_amdgcn_wmma_f32_16x16x32_bf16(false, a, false, b,
                                                 (short)0, c, false, false);
}

// ---------------------------------------------------------------------------
// Kernel 0: zero the gated-output accumulator and per-expert counts.
// ---------------------------------------------------------------------------
__global__ void zero_kernel(float* out_accum, int* counts) {
  int i = blockIdx.x * 256 + threadIdx.x;
  if (i < NELEM) out_accum[i] = 0.0f;
  if (i < NEXP)  counts[i] = 0;
}

// ---------------------------------------------------------------------------
// Kernel 1: repack weights (f32 [E][K][N]) into bf16 WMMA B-fragment layout:
// [e][n_tile][k_blk][lane][16]; lane holds column n = nt*16 + (lane&15) and
// 16 consecutive k values k = kb*32 + 16*(lane>>4) + elem, so a fragment load
// is 32 contiguous bytes per lane (coalesced global_load_b128 pairs).
// ---------------------------------------------------------------------------
__global__ void pack_w_kernel(const float* __restrict__ w, __bf16* __restrict__ wp,
                              int K, int N, int total) {
  int idx = blockIdx.x * 256 + threadIdx.x;
  if (idx >= total) return;
  int elem = idx & 15;
  int lane = (idx >> 4) & 31;
  int kb_cnt = K >> 5;
  int nt_cnt = N >> 4;
  int kb   = (idx >> 9) % kb_cnt;
  int rest = (idx >> 9) / kb_cnt;
  int nt   = rest % nt_cnt;
  int e    = rest / nt_cnt;
  int k = kb * 32 + ((lane >> 4) << 4) + elem;
  int n = nt * 16 + (lane & 15);
  wp[idx] = (__bf16)w[((size_t)e * K + k) * N + n];
}

// ---------------------------------------------------------------------------
// Kernel 2: router. One wave per token: logits = x@gate_w + gate_b, top-2 +
// softmax, atomic scatter into per-expert token/gate lists; emits bf16 x.
// ---------------------------------------------------------------------------
__global__ __launch_bounds__(256) void gate_kernel(
    const float* __restrict__ x, const float* __restrict__ gate_w,
    const float* __restrict__ gate_b, __bf16* __restrict__ xb,
    int* __restrict__ counts, int* __restrict__ etok, float* __restrict__ egate) {
  int lane = threadIdx.x & 31;
  int wid  = threadIdx.x >> 5;
  int t = blockIdx.x * 8 + wid;           // 512 blocks * 8 waves = 4096 tokens

  float acc[NEXP];
#pragma unroll
  for (int e = 0; e < NEXP; ++e) acc[e] = 0.0f;

  for (int d = lane; d < DDIM; d += 32) {
    float xv = x[(size_t)t * DDIM + d];
    xb[(size_t)t * DDIM + d] = (__bf16)xv;
#pragma unroll
    for (int e = 0; e < NEXP; ++e) acc[e] += xv * gate_w[d * NEXP + e];
  }
#pragma unroll
  for (int off = 16; off > 0; off >>= 1) {
#pragma unroll
    for (int e = 0; e < NEXP; ++e) acc[e] += __shfl_xor(acc[e], off, 32);
  }

  if (lane == 0) {
    float l[NEXP];
#pragma unroll
    for (int e = 0; e < NEXP; ++e) l[e] = acc[e] + gate_b[e];
    int i0 = 0; float v0 = l[0];
#pragma unroll
    for (int e = 1; e < NEXP; ++e) if (l[e] > v0) { v0 = l[e]; i0 = e; }
    int i1 = (i0 == 0) ? 1 : 0; float v1 = l[i1];
#pragma unroll
    for (int e = 0; e < NEXP; ++e)
      if (e != i0 && l[e] > v1) { v1 = l[e]; i1 = e; }
    float e1 = __expf(v1 - v0);           // softmax over {v0 (max), v1}
    float g0 = 1.0f / (1.0f + e1);
    float g1 = e1 * g0;
    int s0 = atomicAdd(&counts[i0], 1);
    etok[i0 * N_TOK + s0] = t; egate[i0 * N_TOK + s0] = g0;
    int s1 = atomicAdd(&counts[i1], 1);
    etok[i1 * N_TOK + s1] = t; egate[i1 * N_TOK + s1] = g1;
  }
}

// ---------------------------------------------------------------------------
// Kernel 3: routed fused FFN. Grid = E * 128 tile-slots; each block handles a
// 32-token tile of one expert's token list with 8 waves (256 threads).
// Per H-chunk of 256:
//   GEMM1: wave owns 2 h n-tiles; 2 A frags (row halves) share each B frag.
//   GEMM2: wave owns 4 D n-tiles x 2 row halves (8 persistent accumulators).
// Gate scaling + b2 applied at the end; f32 atomic scatter-add per token.
// ---------------------------------------------------------------------------
__global__ __launch_bounds__(256) void ffn_kernel(
    const __bf16* __restrict__ xb, const __bf16* __restrict__ w1p,
    const __bf16* __restrict__ w2p, const float* __restrict__ b1,
    const float* __restrict__ b2, const int* __restrict__ counts,
    const int* __restrict__ etok, const float* __restrict__ egate,
    float* __restrict__ out_accum) {
  const int e    = blockIdx.x >> 7;      // expert
  const int tile = blockIdx.x & 127;     // tile within expert's list
  const int cnt  = counts[e];
  if (tile * MT >= cnt) return;          // uniform per block: EXEC stays all-1s

  __shared__ __bf16 Xs[MT][XS_STR];      // 32 x 520 bf16  (~33.3 KB)
  __shared__ __bf16 Hs[MT][HS_STR];      // 32 x 264 bf16  (~16.9 KB)
  __shared__ int   toks[MT];
  __shared__ float gs[MT];

  const int tid  = threadIdx.x;
  const int lane = tid & 31;
  const int wv   = tid >> 5;             // wave 0..7
  const int half = lane >> 4;
  const int nlo  = lane & 15;

  if (tid < MT) {
    int slot = tile * MT + tid;
    toks[tid] = (slot < cnt) ? etok[e * N_TOK + slot] : 0;
    gs[tid]   = (slot < cnt) ? egate[e * N_TOK + slot] : 0.0f;
  }
  __syncthreads();

  // Cooperative gather of the 32 token rows (int4 = 8 bf16 per transfer).
  for (int idx = tid; idx < MT * 64; idx += 256) {
    int row = idx >> 6, c8 = idx & 63;
    *(int4*)&Xs[row][c8 * 8] =
        *(const int4*)&xb[(size_t)toks[row] * DDIM + c8 * 8];
  }
  __syncthreads();

  v8f acc2[2][4];                        // [row-half][D n-tile]
#pragma unroll
  for (int mh = 0; mh < 2; ++mh)
#pragma unroll
    for (int i = 0; i < 4; ++i)
#pragma unroll
      for (int r = 0; r < 8; ++r) acc2[mh][i][r] = 0.0f;

  for (int c = 0; c < HDIM / HCHUNK; ++c) {      // 8 chunks of 256
    // ---- GEMM1: h n-tiles {c*16 + 2*wv, +1}, K over D=512 ----
    v8f acc1[2][2];
#pragma unroll
    for (int mh = 0; mh < 2; ++mh)
#pragma unroll
      for (int i = 0; i < 2; ++i)
#pragma unroll
        for (int r = 0; r < 8; ++r) acc1[mh][i][r] = 0.0f;

    for (int kb = 0; kb < DDIM / 32; ++kb) {     // 16 k-steps
      v16bf alo = load_a16(&Xs[0][0], XS_STR, nlo,      kb * 32, half);
      v16bf ahi = load_a16(&Xs[0][0], XS_STR, 16 + nlo, kb * 32, half);
#pragma unroll
      for (int i = 0; i < 2; ++i) {
        int nt = c * 16 + 2 * wv + i;            // 0..127 (H/16)
        size_t off = ((((size_t)e * 128 + nt) * 16 + kb) * 32 + lane) * 16;
        v16bf bf = *(const v16bf*)(w1p + off);
        acc1[0][i] = wmma_bf16(alo, bf, acc1[0][i]);
        acc1[1][i] = wmma_bf16(ahi, bf, acc1[1][i]);
      }
    }
    // bias + relu -> bf16 h chunk in LDS
#pragma unroll
    for (int mh = 0; mh < 2; ++mh) {
#pragma unroll
      for (int i = 0; i < 2; ++i) {
        int coll = (2 * wv + i) * 16 + nlo;      // col within chunk [0,256)
        int colg = c * HCHUNK + coll;            // global h col
        float bias = b1[e * HDIM + colg];
#pragma unroll
        for (int r = 0; r < 8; ++r) {
          int m2 = mh * 16 + r + 8 * half;
          float hv = fmaxf(acc1[mh][i][r] + bias, 0.0f);
          Hs[m2][coll] = (__bf16)hv;
        }
      }
    }
    __syncthreads();

    // ---- GEMM2 partial: wave owns D n-tiles 4*wv..4*wv+3, K = this chunk ----
    for (int kb = 0; kb < HCHUNK / 32; ++kb) {   // 8 k-steps
      v16bf alo = load_a16(&Hs[0][0], HS_STR, nlo,      kb * 32, half);
      v16bf ahi = load_a16(&Hs[0][0], HS_STR, 16 + nlo, kb * 32, half);
#pragma unroll
      for (int i = 0; i < 4; ++i) {
        int nt  = 4 * wv + i;                    // 0..31 (D/16)
        int kbg = c * (HCHUNK / 32) + kb;        // 0..63 (H/32)
        size_t off = ((((size_t)e * 32 + nt) * 64 + kbg) * 32 + lane) * 16;
        v16bf bf = *(const v16bf*)(w2p + off);
        acc2[0][i] = wmma_bf16(alo, bf, acc2[0][i]);
        acc2[1][i] = wmma_bf16(ahi, bf, acc2[1][i]);
      }
    }
    __syncthreads();                             // Hs reused next chunk
  }

  // gate-scale (+b2) and scatter-add per token
#pragma unroll
  for (int mh = 0; mh < 2; ++mh) {
#pragma unroll
    for (int i = 0; i < 4; ++i) {
      int n = (4 * wv + i) * 16 + nlo;
      float bias = b2[e * DDIM + n];
#pragma unroll
      for (int r = 0; r < 8; ++r) {
        int m2 = mh * 16 + r + 8 * half;
        float val = (acc2[mh][i][r] + bias) * gs[m2];
        atomicAdd(&out_accum[(size_t)toks[m2] * DDIM + n], val);
      }
    }
  }
}

// ---------------------------------------------------------------------------
// Kernel 4: Adam-style update + LayerNorm, one block (512 thr) per token.
// Outputs tuple (out, p_new, v_new, momentum) concatenated in d_out.
// ---------------------------------------------------------------------------
__global__ __launch_bounds__(512) void finalize_kernel(
    const float* __restrict__ x, const float* __restrict__ p,
    const float* __restrict__ v, const float* __restrict__ m,
    const float* __restrict__ ln_w, const float* __restrict__ ln_b,
    const float* __restrict__ out_accum, float* __restrict__ out) {
  __shared__ float red1[16], red2[16];
  int t = blockIdx.x;
  int d = threadIdx.x;
  int lane = d & 31, wid = d >> 5;
  size_t i = (size_t)t * DDIM + d;

  float eo = out_accum[i];
  float mom = MU_C * m[i] + G2_C * eo;
  float pn  = BETA1_C * p[i] + (1.0f - BETA1_C) * eo;
  float vn  = BETA2_C * v[i] + (1.0f - BETA2_C) * eo * eo;
  float adam_m_x = G1_C * rsqrtf(vn + EPSA_C) * pn;  // adam - x
  float y = -adam_m_x;                               // x - adam

  float s = y, s2 = y * y;
#pragma unroll
  for (int off = 16; off > 0; off >>= 1) {
    s  += __shfl_xor(s,  off, 32);
    s2 += __shfl_xor(s2, off, 32);
  }
  if (lane == 0) { red1[wid] = s; red2[wid] = s2; }
  __syncthreads();
  if (wid == 0) {
    float a = (lane < 16) ? red1[lane] : 0.0f;
    float b = (lane < 16) ? red2[lane] : 0.0f;
#pragma unroll
    for (int off = 16; off > 0; off >>= 1) {
      a += __shfl_xor(a, off, 32);
      b += __shfl_xor(b, off, 32);
    }
    if (lane == 0) { red1[0] = a; red2[0] = b; }
  }
  __syncthreads();
  float mu  = red1[0] * (1.0f / DDIM);
  float var = red2[0] * (1.0f / DDIM) - mu * mu;
  float yn  = (y - mu) * rsqrtf(var + EPSLN_C) * ln_w[d] + ln_b[d];

  out[i]             = yn;
  out[NELEM + i]     = pn;
  out[2 * NELEM + i] = vn;
  out[3 * NELEM + i] = mom;
}

// ---------------------------------------------------------------------------
extern "C" void kernel_launch(void* const* d_in, const int* in_sizes, int n_in,
                              void* d_out, int out_size, void* d_ws, size_t ws_size,
                              hipStream_t stream) {
  const float* x      = (const float*)d_in[0];
  const float* p      = (const float*)d_in[1];
  const float* v      = (const float*)d_in[2];
  const float* m      = (const float*)d_in[3];
  const float* gate_w = (const float*)d_in[4];
  const float* gate_b = (const float*)d_in[5];
  const float* w1     = (const float*)d_in[6];
  const float* b1     = (const float*)d_in[7];
  const float* w2     = (const float*)d_in[8];
  const float* b2     = (const float*)d_in[9];
  const float* ln_w   = (const float*)d_in[10];
  const float* ln_b   = (const float*)d_in[11];
  float* out = (float*)d_out;

  // Workspace carve-up (~44.3 MiB total)
  char* wsb = (char*)d_ws;
  float*  out_accum = (float*)wsb;                                // 8 MiB
  int*    counts    = (int*)(wsb + (size_t)8 * 1024 * 1024);      // 8 ints (pad 64)
  int*    etok      = counts + 64;                                // 8*4096 ints
  float*  egate     = (float*)(etok + NEXP * N_TOK);              // 8*4096 f32
  __bf16* xb        = (__bf16*)(egate + NEXP * N_TOK);            // 4 MiB
  __bf16* w1p       = xb + (size_t)N_TOK * DDIM;                  // 16 MiB
  __bf16* w2p       = w1p + (size_t)NEXP * DDIM * HDIM;           // 16 MiB

  const int wtotal = NEXP * DDIM * HDIM;  // 8,388,608 packed elements each

  zero_kernel<<<NELEM / 256, 256, 0, stream>>>(out_accum, counts);
  pack_w_kernel<<<wtotal / 256, 256, 0, stream>>>(w1, w1p, DDIM, HDIM, wtotal);
  pack_w_kernel<<<wtotal / 256, 256, 0, stream>>>(w2, w2p, HDIM, DDIM, wtotal);
  gate_kernel<<<N_TOK / 8, 256, 0, stream>>>(x, gate_w, gate_b, xb,
                                             counts, etok, egate);
  ffn_kernel<<<NEXP * TILES, 256, 0, stream>>>(xb, w1p, w2p, b1, b2,
                                               counts, etok, egate, out_accum);
  finalize_kernel<<<N_TOK, 512, 0, stream>>>(x, p, v, m, ln_w, ln_b,
                                             out_accum, out);
}
